// MultiHeadAttention_9938554323595
// MI455X (gfx1250) — matrix operand order repaired and checked by reading, not compiled, not run on hardware
//
#include <hip/hip_runtime.h>
#include <hip/hip_bf16.h>
#include <math.h>

// ---------------------------------------------------------------------------
// MI455X fused multi-head attention (compile-only target: gfx1250, wave32)
//   H=8, DK=64, DIN=512, B=64, S=512, DV=64
// All GEMMs via v_wmma_f32_16x16x32_bf16 (f32 accumulate).
// All WMMA operand loads are contiguous global_load_b128 (V is pre-transposed).
// ---------------------------------------------------------------------------

typedef __bf16 v16bf __attribute__((ext_vector_type(16)));
typedef __bf16 v8bf  __attribute__((ext_vector_type(8)));
typedef __bf16 v4bf  __attribute__((ext_vector_type(4)));
typedef float  v8f   __attribute__((ext_vector_type(8)));

constexpr int cH   = 8;
constexpr int cDK  = 64;
constexpr int cDIN = 512;
constexpr int cB   = 64;
constexpr int cS   = 512;
constexpr int cDV  = 64;
constexpr int cBS  = cB * cS;          // 32768
constexpr float INV_TEMP  = 0.125f;    // 1/sqrt(64)
constexpr float MASK_FILL = -1000.0f;

static __device__ inline v8f zero8() {
  v8f z = {0.f, 0.f, 0.f, 0.f, 0.f, 0.f, 0.f, 0.f};
  return z;
}

static __device__ inline v8f wmma_bf16(v16bf a, v16bf b, v8f c) {
  // (neg_a, A, neg_b, B, c_mod, C, reuse_a, reuse_b)
  return __builtin_amdgcn_wmma_f32_16x16x32_bf16(false, a, false, b, (short)0, c,
                                                 false, false);
}

// A-matrix 16x32 bf16 fragment from a row-major [rows][ld] source.
// ISA layout: lane l -> row = row0 + (l&15); koff = (l>>4)*8;
// elems 0..7 = K k0+koff+0..7, elems 8..15 = K k0+16+koff+0..7.
static __device__ inline v16bf load_a_frag(const __bf16* __restrict__ base, int ld,
                                           int row0, int k0) {
  const int l = threadIdx.x & 31;
  const __bf16* p = base + (size_t)(row0 + (l & 15)) * ld + k0 + ((l >> 4) ? 8 : 0);
  v8bf lo = *(const v8bf*)p;        // global_load_b128
  v8bf hi = *(const v8bf*)(p + 16); // global_load_b128
  v16bf a;
#pragma unroll
  for (int t = 0; t < 8; ++t) { a[t] = lo[t]; a[t + 8] = hi[t]; }
  return a;
}

// B-matrix 32x16 bf16 fragment where B[k][n] = src[n0+n][k0+k]: columns of B
// are contiguous source rows (used for Q^T, W^T, and transposed-V).
// ISA layout: lane l -> n = l&15; koff = (l>>4)*8;
// elems 0..7 = K k0+koff+0..7, elems 8..15 = K k0+16+koff+0..7.
static __device__ inline v16bf load_b_frag_rows(const __bf16* __restrict__ base, int ld,
                                                int n0, int k0) {
  const int l = threadIdx.x & 31;
  const __bf16* p = base + (size_t)(n0 + (l & 15)) * ld + k0 + ((l >> 4) ? 8 : 0);
  v8bf lo = *(const v8bf*)p;
  v8bf hi = *(const v8bf*)(p + 16);
  v16bf b;
#pragma unroll
  for (int t = 0; t < 8; ++t) { b[t] = lo[t]; b[t + 8] = hi[t]; }
  return b;
}

// ---------------------------------------------------------------------------
// Kernel 0: vectorized f32 -> bf16 (v, Wq, Wk). n must be a multiple of 4.
// ---------------------------------------------------------------------------
__global__ void cvt_f32_bf16(const float* __restrict__ src, __bf16* __restrict__ dst,
                             int n4) {
  int i = blockIdx.x * blockDim.x + threadIdx.x;
  const int stride = gridDim.x * blockDim.x;
  for (; i < n4; i += stride) {
    float4 f = ((const float4*)src)[i];
    v4bf o = {(__bf16)f.x, (__bf16)f.y, (__bf16)f.z, (__bf16)f.w};
    ((v4bf*)dst)[i] = o;
  }
}

// ---------------------------------------------------------------------------
// Kernel 0b: per-batch transpose of v: (B,S,DIN) f32 -> vT (B,DIN,S) bf16.
// Makes V columns contiguous so the P@V B-operand is loaded with b128s.
// 32x32 f32 LDS tiles; grid (S/32, DIN/32, B), block (32,8).
// ---------------------------------------------------------------------------
__global__ void __launch_bounds__(256) transpose_v_kernel(
    const float* __restrict__ v, __bf16* __restrict__ vT) {
  __shared__ float tile[32][33];
  const int b  = blockIdx.z;
  const int s0 = blockIdx.x * 32;
  const int d0 = blockIdx.y * 32;
#pragma unroll
  for (int yy = threadIdx.y; yy < 32; yy += 8)
    tile[yy][threadIdx.x] =
        v[((size_t)b * cS + s0 + yy) * cDIN + d0 + threadIdx.x];
  __syncthreads();
#pragma unroll
  for (int yy = threadIdx.y; yy < 32; yy += 8)
    vT[((size_t)b * cDIN + d0 + yy) * cS + s0 + threadIdx.x] =
        (__bf16)tile[threadIdx.x][yy];
}

// ---------------------------------------------------------------------------
// Kernel 1: Q/K projection GEMM.  C[m][e] = sum_d v[m][d] * W[e][d] + bias[e]
// m over BS=32768, e over 512.  One wave per 16x16 C tile, K-loop 512/32.
// grid = (BS/16, 32/8, 2[q|k]), block = 256 (8 waves).
// ---------------------------------------------------------------------------
__global__ void __launch_bounds__(256) proj_kernel(
    const __bf16* __restrict__ vb,
    const __bf16* __restrict__ wqb, const float* __restrict__ bq,
    __bf16* __restrict__ qb,
    const __bf16* __restrict__ wkb, const float* __restrict__ bk,
    __bf16* __restrict__ kbuf) {
  const __bf16* wb   = blockIdx.z ? wkb : wqb;
  const float*  bias = blockIdx.z ? bk  : bq;
  __bf16*       dst  = blockIdx.z ? kbuf : qb;

  const int wave = threadIdx.x >> 5;
  const int l    = threadIdx.x & 31;
  const int half = l >> 4;
  const int mt   = blockIdx.x * 16;
  const int nt   = (blockIdx.y * 8 + wave) * 16;

  v8f acc = zero8();
#pragma unroll
  for (int kk = 0; kk < cDIN; kk += 32) {
    v16bf a = load_a_frag(vb, cDIN, mt, kk);       // v rows
    v16bf b = load_b_frag_rows(wb, cDIN, nt, kk);  // W rows as B columns
    acc = wmma_bf16(a, b, acc);
  }
  const float be = bias[nt + (l & 15)];
#pragma unroll
  for (int r = 0; r < 8; ++r) {
    float cv = acc[r] + be;
    dst[(size_t)(mt + r + 8 * half) * cDIN + nt + (l & 15)] = (__bf16)cv;
  }
}

// ---------------------------------------------------------------------------
// Kernel 2: fused attention.  One wave owns a 16-row strip of attn for one
// (h,b): logits (64 WMMA) -> register-resident softmax (32 v8f accumulators
// = 256 VGPRs/lane stay live; CDNA5 allows 1024) -> single attn HBM write ->
// P@Vh (64 WMMA) with a 16x32 LDS relayout of P (C-layout -> A-layout).
// grid = (S/64, B, H), block = 128 (4 waves).
// ---------------------------------------------------------------------------
__global__ void __launch_bounds__(128) attn_kernel(
    const __bf16* __restrict__ qb, const __bf16* __restrict__ kb,
    const __bf16* __restrict__ vT, const unsigned char* __restrict__ pad,
    float* __restrict__ out, float* __restrict__ attn) {
  __shared__ float ldsT[4][16][33];  // per-wave 16x32 P-tile relayout (+pad)

  const int wave = threadIdx.x >> 5;
  const int l    = threadIdx.x & 31;
  const int half = l >> 4;
  const int n    = l & 15;
  const int h    = blockIdx.z;
  const int b    = blockIdx.y;
  const int ibase = blockIdx.x * 64 + wave * 16;

  const __bf16* qrow  = qb + (size_t)b * cS * cDIN + h * cDK;   // q[j][d], ld 512
  const __bf16* krow  = kb + (size_t)b * cS * cDIN + h * cDK;   // k[i][d], ld 512
  const __bf16* vTrow = vT + ((size_t)b * cDIN + h * cDV) * cS; // vT[dv][j], ld 512

  // K-operand fragments for this strip's 16 rows (K = DK = 64): reused for all
  // 32 j-tiles.
  const v16bf ka0 = load_a_frag(krow, cDIN, ibase, 0);
  const v16bf ka1 = load_a_frag(krow, cDIN, ibase, 32);

  // ---- logits: attn[i][j] = (k_i . q_j) / temp, masked on j --------------
  v8f acc[32];
#pragma unroll
  for (int jt = 0; jt < 32; ++jt) {
    if (jt + 1 < 32)  // global_prefetch_b8 of next Q tile
      __builtin_prefetch(qrow + (size_t)((jt + 1) * 16 + n) * cDIN, 0, 3);
    v16bf b0 = load_b_frag_rows(qrow, cDIN, jt * 16, 0);
    v16bf b1 = load_b_frag_rows(qrow, cDIN, jt * 16, 32);
    v8f c = zero8();
    c = wmma_bf16(ka0, b0, c);
    c = wmma_bf16(ka1, b1, c);
    const bool mk = pad[b * cS + jt * 16 + n] != 0;  // lane's column j
#pragma unroll
    for (int r = 0; r < 8; ++r)
      acc[jt][r] = mk ? MASK_FILL : c[r] * INV_TEMP;
  }

  // ---- softmax over j, fully in registers --------------------------------
  // Row m = r + 8*half lives in one 16-lane half: reduce with xor 1,2,4,8.
#pragma unroll
  for (int r = 0; r < 8; ++r) {
    float m = -3.4e38f;
#pragma unroll
    for (int jt = 0; jt < 32; ++jt) m = fmaxf(m, acc[jt][r]);
    m = fmaxf(m, __shfl_xor(m, 1, 32));
    m = fmaxf(m, __shfl_xor(m, 2, 32));
    m = fmaxf(m, __shfl_xor(m, 4, 32));
    m = fmaxf(m, __shfl_xor(m, 8, 32));
    float s = 0.f;
#pragma unroll
    for (int jt = 0; jt < 32; ++jt) {
      float e = __expf(acc[jt][r] - m);
      acc[jt][r] = e;
      s += e;
    }
    s += __shfl_xor(s, 1, 32);
    s += __shfl_xor(s, 2, 32);
    s += __shfl_xor(s, 4, 32);
    s += __shfl_xor(s, 8, 32);
    const float inv = 1.f / s;
#pragma unroll
    for (int jt = 0; jt < 32; ++jt) acc[jt][r] *= inv;
  }

  // ---- store attn (the single HBM write of the 537MB tensor) -------------
  float* arow = attn + (((size_t)h * cB + b) * cS + ibase) * cS;
#pragma unroll
  for (int r = 0; r < 8; ++r) {
    const int row = r + 8 * half;
#pragma unroll
    for (int jt = 0; jt < 32; ++jt)
      arow[(size_t)row * cS + jt * 16 + n] = acc[jt][r];
  }

  // ---- out = P @ Vh  (16x512 @ 512x64), Vh columns contiguous via vT -----
  v8f oacc[4] = {zero8(), zero8(), zero8(), zero8()};
#pragma unroll
  for (int kk = 0; kk < 16; ++kk) {  // 32 P-columns per step
    if (kk + 1 < 16)  // prefetch next V^T stripe
      __builtin_prefetch(vTrow + (size_t)n * cS + (kk + 1) * 32, 0, 3);
    // Stage normalized P tile (C-layout regs -> LDS) for A-layout reread.
#pragma unroll
    for (int r = 0; r < 8; ++r) {
      ldsT[wave][r + 8 * half][n]      = acc[2 * kk][r];
      ldsT[wave][r + 8 * half][16 + n] = acc[2 * kk + 1][r];
    }
    // In-wave LDS is processed in order; wait + memory clobber keeps the
    // compiler from sinking the cross-lane reads above the writes.
    asm volatile("s_wait_dscnt 0x0" ::: "memory");
    v16bf pa;
    {
      const int row  = l & 15;
      const int koff = half * 8;
#pragma unroll
      for (int t = 0; t < 8; ++t) {
        pa[t]     = (__bf16)ldsT[wave][row][koff + t];
        pa[t + 8] = (__bf16)ldsT[wave][row][16 + koff + t];
      }
    }
#pragma unroll
    for (int ntl = 0; ntl < 4; ++ntl) {
      // B[k][n] = vh[kk*32+k][ntl*16+n] = vTrow[ntl*16+n][kk*32+k]
      v16bf bv = load_b_frag_rows(vTrow, cS, ntl * 16, kk * 32);
      oacc[ntl] = wmma_bf16(pa, bv, oacc[ntl]);
    }
  }

  float* orow = out + (((size_t)h * cB + b) * cS + ibase) * cDV;
#pragma unroll
  for (int ntl = 0; ntl < 4; ++ntl)
#pragma unroll
    for (int r = 0; r < 8; ++r)
      orow[(size_t)(r + 8 * half) * cDV + ntl * 16 + n] = oacc[ntl][r];
}

// ---------------------------------------------------------------------------
extern "C" void kernel_launch(void* const* d_in, const int* in_sizes, int n_in,
                              void* d_out, int out_size, void* d_ws, size_t ws_size,
                              hipStream_t stream) {
  const float* v_f32  = (const float*)d_in[0];
  const float* Wq_f32 = (const float*)d_in[1];
  const float* bq     = (const float*)d_in[2];
  const float* Wk_f32 = (const float*)d_in[3];
  const float* bk     = (const float*)d_in[4];
  const unsigned char* pad = (const unsigned char*)d_in[5];  // bool mask (B,S)

  // Workspace layout (bf16):
  //   vb  : BS*DIN   (33.5 MB)  -- row-major v (proj A operand)
  //   qb  : BS*DIN   (33.5 MB)
  //   kb  : BS*DIN   (33.5 MB)
  //   vT  : BS*DIN   (33.5 MB)  -- per-batch transposed v (P@V B operand)
  //   wqb : DIN*DIN  (0.5 MB)
  //   wkb : DIN*DIN  (0.5 MB)
  __bf16* vb  = (__bf16*)d_ws;
  __bf16* qb  = vb  + (size_t)cBS * cDIN;
  __bf16* kb  = qb  + (size_t)cBS * cDIN;
  __bf16* vT  = kb  + (size_t)cBS * cDIN;
  __bf16* wqb = vT  + (size_t)cBS * cDIN;
  __bf16* wkb = wqb + (size_t)cDIN * cDIN;

  float* out  = (float*)d_out;                        // (H,B,S,DV)
  float* attn = out + (size_t)cH * cB * cS * cDV;     // (H,B,S,S)

  // 0) bf16 conversions (vectorized) + V transpose
  cvt_f32_bf16<<<dim3(1024), dim3(256), 0, stream>>>(v_f32, vb, cBS * cDIN / 4);
  cvt_f32_bf16<<<dim3(64), dim3(256), 0, stream>>>(Wq_f32, wqb, cDIN * cDIN / 4);
  cvt_f32_bf16<<<dim3(64), dim3(256), 0, stream>>>(Wk_f32, wkb, cDIN * cDIN / 4);
  transpose_v_kernel<<<dim3(cS / 32, cDIN / 32, cB), dim3(32, 8), 0, stream>>>(
      v_f32, vT);

  // 1) Q/K projections: (BS x 512) = (BS x 512) * (512 x 512)^T, bias add
  proj_kernel<<<dim3(cBS / 16, (cDIN / 16) / 8, 2), dim3(256), 0, stream>>>(
      vb, wqb, bq, qb, wkb, bk, kb);

  // 2) fused logits + softmax + attn store + P@Vh
  attn_kernel<<<dim3(cS / 64, cB, cH), dim3(128), 0, stream>>>(
      qb, kb, vT, pad, out, attn);
}
